// RNN_Multi_Head_65747359367446
// MI455X (gfx1250) — matrix operand order repaired and checked by reading, compiled, not run on hardware
//
#include <hip/hip_runtime.h>
#include <cstdint>
#include <cstddef>

// ---------------------------------------------------------------------------
// MI455X (gfx1250) implementation of the multi-head GRU stack.
//  - bf16 WMMA (v_wmma_f32_16x16x32_bf16) for all GEMMs and the recurrent
//    per-step matmuls; f32 accumulation.
//  - Persistent per-head scan kernels: weights + hidden state resident in LDS
//    across all 512 timesteps (two workgroup barriers per step).
//  - Scalar-recurrence kernel for the hid=1 block (b3).
//  - Branch-free float4 (b128) global operand loads; K-aligned + RELU
//    specializations of the GEMM are template-dispatched so the dominant
//    paths carry no tail selects and no runtime-flag epilogue.
// ---------------------------------------------------------------------------

typedef __attribute__((ext_vector_type(16))) __bf16 v16bf;
typedef __attribute__((ext_vector_type(8)))  __bf16 v8bf;
typedef __attribute__((ext_vector_type(4)))  __bf16 v4bf;
typedef __attribute__((ext_vector_type(2)))  __bf16 v2bf;
typedef __attribute__((ext_vector_type(8)))  float  v8f;

__device__ inline v8f wmma_bf16(v16bf a, v16bf b, v8f c) {
#if defined(__HIP_DEVICE_COMPILE__)
  return __builtin_amdgcn_wmma_f32_16x16x32_bf16(false, a, false, b, (short)0, c,
                                                 false, false);
#else
  return c;
#endif
}

__device__ inline float sigf(float x) { return 1.0f / (1.0f + __expf(-x)); }

// ---- LDS tile gathers (contiguous k = 16*(lane/16) + s, same order in A and B) ---------------------------
// A-tile (16x32 bf16, MxK): lane%16 = M row, lane/16 selects K halves,
// adjacent-K pairs packed -> read pairs as v2bf (ds_load_b32).
__device__ inline v16bf ldsA(const __bf16* base, int stride, int kb, int lane) {
  const int m = lane & 15, g = lane >> 4;
  const __bf16* row = base + (size_t)m * stride + kb;
  v16bf a;
#pragma unroll
  for (int v = 0; v < 8; ++v) {
    const int k = g * 16 + 2 * v;
    const v2bf p = *(const v2bf*)(row + k);
    a[2 * v] = p.x;
    a[2 * v + 1] = p.y;
  }
  return a;
}

// B-tile (32x16 bf16, KxN) from row-major W[N][K]: lane%16 = N column,
// lane/16 selects K group of 16 -> 16 contiguous bf16 = two v8bf loads.
__device__ inline v16bf ldsB(const __bf16* base, int stride, int kb, int lane) {
  const int n = lane & 15, g = lane >> 4;
  const __bf16* row = base + (size_t)n * stride + kb + g * 16;
  const v8bf lo = *(const v8bf*)(row);
  const v8bf hi = *(const v8bf*)(row + 8);
  v16bf b;
#pragma unroll
  for (int i = 0; i < 8; ++i) { b[i] = lo[i]; b[8 + i] = hi[i]; }
  return b;
}

// ---- Global operand gathers for the generic GEMM --------------------------
// Requirements: M % 16 == 0 (all call sites), K % 4 == 0, A/W 16B aligned.
// KALIGNED: K % 32 == 0 -> pure float4 loads.
// Otherwise the K tail is handled branch-free: clamped float4 + select-zero.
template <bool KALIGNED>
__device__ inline v16bf gblA(const float* __restrict__ A, int K, int mt, int kb,
                             int lane) {
  const int m = mt * 16 + (lane & 15);
  const int g = lane >> 4;
  const float* row = A + (size_t)m * K;
  v16bf a;
#pragma unroll
  for (int c = 0; c < 4; ++c) {
    const int k0 = kb + g * 16 + 4 * c;
    if (KALIGNED) {
      const float4 p = *(const float4*)(row + k0);
      a[4 * c + 0] = (__bf16)p.x;
      a[4 * c + 1] = (__bf16)p.y;
      a[4 * c + 2] = (__bf16)p.z;
      a[4 * c + 3] = (__bf16)p.w;
    } else {
      const int kc = (k0 + 4 <= K) ? k0 : (K - 4);
      const float4 p = *(const float4*)(row + kc);
      const bool ok = (k0 + 4 <= K);
      a[4 * c + 0] = (__bf16)(ok ? p.x : 0.0f);
      a[4 * c + 1] = (__bf16)(ok ? p.y : 0.0f);
      a[4 * c + 2] = (__bf16)(ok ? p.z : 0.0f);
      a[4 * c + 3] = (__bf16)(ok ? p.w : 0.0f);
    }
  }
  return a;
}

// Out-of-range N rows feed columns that are never stored -> clamp row index.
template <bool KALIGNED>
__device__ inline v16bf gblB(const float* __restrict__ W, int N, int K, int nt,
                             int kb, int lane) {
  int n = nt * 16 + (lane & 15);
  n = (n < N) ? n : (N - 1);
  const int g = lane >> 4;
  const float* row = W + (size_t)n * K;
  v16bf b;
#pragma unroll
  for (int c = 0; c < 4; ++c) {
    const int k0 = kb + g * 16 + 4 * c;
    if (KALIGNED) {
      const float4 p = *(const float4*)(row + k0);
      b[4 * c + 0] = (__bf16)p.x;
      b[4 * c + 1] = (__bf16)p.y;
      b[4 * c + 2] = (__bf16)p.z;
      b[4 * c + 3] = (__bf16)p.w;
    } else {
      const int kc = (k0 + 4 <= K) ? k0 : (K - 4);
      const float4 p = *(const float4*)(row + kc);
      const bool ok = (k0 + 4 <= K);
      b[4 * c + 0] = (__bf16)(ok ? p.x : 0.0f);
      b[4 * c + 1] = (__bf16)(ok ? p.y : 0.0f);
      b[4 * c + 2] = (__bf16)(ok ? p.z : 0.0f);
      b[4 * c + 3] = (__bf16)(ok ? p.w : 0.0f);
    }
  }
  return b;
}

// ---------------------------------------------------------------------------
// Generic batched GEMM: C[g] = act(A[g][M][K] * W[g][N][K]^T + bias[g][N])
// One 16x16 output tile per wave. M must be a multiple of 16.
// ---------------------------------------------------------------------------
template <bool KALIGNED, bool RELU>
__global__ __launch_bounds__(256) void wmma_gemm_kernel(
    const float* __restrict__ A, const float* __restrict__ W,
    const float* __restrict__ bias, float* __restrict__ C,
    int M, int N, int K, int MT, int NT,
    size_t sA, size_t sW, size_t sB, size_t sC, long tiles) {
  const long wid = (long)blockIdx.x * 8 + (threadIdx.x >> 5);
  if (wid >= tiles) return;
  const int lane = threadIdx.x & 31;
  const long per = (long)MT * NT;
  const int g = (int)(wid / per);
  const long r = wid % per;
  const int mt = (int)(r / NT), nt = (int)(r % NT);
  const float* Ag = A + (size_t)g * sA;
  const float* Wg = W + (size_t)g * sW;
  const float* bg = bias + (size_t)g * sB;
  const int n = nt * 16 + (lane & 15);
  const float bv = (n < N) ? bg[n] : 0.0f;
  v8f acc;
#pragma unroll
  for (int q = 0; q < 8; ++q) acc[q] = bv;
  for (int kb = 0; kb < K; kb += 32) {
    const v16bf a = gblA<KALIGNED>(Ag, K, mt, kb, lane);
    const v16bf b = gblB<KALIGNED>(Wg, N, K, nt, kb, lane);
    acc = wmma_bf16(a, b, acc);
  }
  float* Cg = C + (size_t)g * sC;
  const int mb = mt * 16 + (lane >> 4) * 8;
  if (n < N) {
#pragma unroll
    for (int q = 0; q < 8; ++q) {
      float v = acc[q];
      if (RELU) v = fmaxf(v, 0.0f);
      Cg[(size_t)(mb + q) * N + n] = v;
    }
  }
}

// ---------------------------------------------------------------------------
// Persistent GRU scan. One workgroup (8 waves) per head; weights (bf16) and
// hidden state (f32) live in LDS for all T steps. Per step:
//   phase A: stage x_t (f32->bf16, float4 loads) and h into LDS; barrier.
//   phase B: WMMA tiles -> 6 f32 accumulators -> gates -> h_new; out = h+res.
// HID padded to HP (mult of 16) so gate boundaries are tile aligned; padded
// columns have zero weights/biases and provably stay at h == 0.
// ---------------------------------------------------------------------------
template <int IN, int HID>
__global__ __launch_bounds__(256) void gru_scan_kernel(
    const float* __restrict__ xin,    // [heads][128][512][IN]
    const float* __restrict__ resid,  // [heads][128][512][HID]
    float* __restrict__ out,          // [heads][128][512][HID]
    const __bf16* __restrict__ wihP,  // [heads][3*HP][KI]
    const __bf16* __restrict__ whhP,  // [heads][3*HP][KH]
    const float* __restrict__ bihP,   // [heads][3*HP]
    const float* __restrict__ bhhP) { // [heads][3*HP]
  constexpr int HP  = (HID + 15) & ~15;
  constexpr int KI  = (IN + 31) & ~31;
  constexpr int KH  = (HP + 31) & ~31;
  constexpr int NB  = 128, T = 512;
  constexpr int JT  = HP / 16;
  constexpr int SXI = KI + 8;   // multiples of 8 -> 16B-aligned rows
  constexpr int SXH = KH + 8;
  constexpr int IN4 = IN / 4;
  constexpr int HP4 = HP / 4;

  __shared__ __align__(16) __bf16 Wih_s[3 * HP][SXI];
  __shared__ __align__(16) __bf16 Whh_s[3 * HP][SXH];
  __shared__ float Bih_s[3 * HP];
  __shared__ float Bhh_s[3 * HP];
  __shared__ __align__(16) __bf16 Xs[NB][SXI];
  __shared__ __align__(16) __bf16 Hb[NB][SXH];
  __shared__ __align__(16) float  Hf[NB][HP];

  const int head = blockIdx.x;
  const int tid  = threadIdx.x;
  const int lane = tid & 31;
  const int wave = tid >> 5;

  {
    const __bf16* w0 = wihP + (size_t)head * (3 * HP * KI);
    for (int i = tid; i < 3 * HP * KI; i += 256) Wih_s[i / KI][i % KI] = w0[i];
    const __bf16* w1 = whhP + (size_t)head * (3 * HP * KH);
    for (int i = tid; i < 3 * HP * KH; i += 256) Whh_s[i / KH][i % KH] = w1[i];
    const float* bi = bihP + (size_t)head * (3 * HP);
    const float* bh = bhhP + (size_t)head * (3 * HP);
    for (int i = tid; i < 3 * HP; i += 256) { Bih_s[i] = bi[i]; Bhh_s[i] = bh[i]; }
    for (int i = tid; i < NB * HP; i += 256) Hf[i / HP][i % HP] = 0.0f;
    // zero Xs/Hb entirely once: pad columns are never rewritten.
    for (int i = tid; i < NB * SXI; i += 256) (&Xs[0][0])[i] = (__bf16)0.0f;
    for (int i = tid; i < NB * SXH; i += 256) (&Hb[0][0])[i] = (__bf16)0.0f;
  }
  __syncthreads();

  const size_t rowBase = (size_t)head * NB;
  for (int t = 0; t < T; ++t) {
    // ---- phase A: stage x_t and h into bf16 LDS (vectorized) ----
    for (int i = tid; i < NB * IN4; i += 256) {
      const int b = i / IN4, k4 = (i % IN4) * 4;
      const float4 v = *(const float4*)&xin[((rowBase + b) * T + t) * IN + k4];
      v4bf o;
      o.x = (__bf16)v.x; o.y = (__bf16)v.y; o.z = (__bf16)v.z; o.w = (__bf16)v.w;
      *(v4bf*)&Xs[b][k4] = o;
    }
    for (int i = tid; i < NB * HP4; i += 256) {
      const int b = i / HP4, k4 = (i % HP4) * 4;
      const float4 v = *(const float4*)&Hf[b][k4];
      v4bf o;
      o.x = (__bf16)v.x; o.y = (__bf16)v.y; o.z = (__bf16)v.z; o.w = (__bf16)v.w;
      *(v4bf*)&Hb[b][k4] = o;
    }
    if (t + 1 < T) {
      const int b = tid & (NB - 1);
      __builtin_prefetch(&xin[((rowBase + b) * T + (t + 1)) * IN], 0, 1);
    }
    __syncthreads();

    // ---- phase B: WMMA tiles + gates ----
    for (int task = wave; task < 8 * JT; task += 8) {
      const int mt = task & 7;
      const int j  = task >> 3;
      const int nc = j * 16 + (lane & 15);
      v8f ir, iz, in_, hr, hz, hn;
      {
        float v;
        v = Bih_s[0 * HP + nc]; for (int q = 0; q < 8; ++q) ir[q]  = v;
        v = Bih_s[1 * HP + nc]; for (int q = 0; q < 8; ++q) iz[q]  = v;
        v = Bih_s[2 * HP + nc]; for (int q = 0; q < 8; ++q) in_[q] = v;
        v = Bhh_s[0 * HP + nc]; for (int q = 0; q < 8; ++q) hr[q]  = v;
        v = Bhh_s[1 * HP + nc]; for (int q = 0; q < 8; ++q) hz[q]  = v;
        v = Bhh_s[2 * HP + nc]; for (int q = 0; q < 8; ++q) hn[q]  = v;
      }
#pragma unroll
      for (int kb = 0; kb < KI; kb += 32) {
        const v16bf a = ldsA(&Xs[0][0] + (size_t)mt * 16 * SXI, SXI, kb, lane);
        ir  = wmma_bf16(a, ldsB(&Wih_s[0 * HP + j * 16][0], SXI, kb, lane), ir);
        iz  = wmma_bf16(a, ldsB(&Wih_s[1 * HP + j * 16][0], SXI, kb, lane), iz);
        in_ = wmma_bf16(a, ldsB(&Wih_s[2 * HP + j * 16][0], SXI, kb, lane), in_);
      }
#pragma unroll
      for (int kb = 0; kb < KH; kb += 32) {
        const v16bf a = ldsA(&Hb[0][0] + (size_t)mt * 16 * SXH, SXH, kb, lane);
        hr = wmma_bf16(a, ldsB(&Whh_s[0 * HP + j * 16][0], SXH, kb, lane), hr);
        hz = wmma_bf16(a, ldsB(&Whh_s[1 * HP + j * 16][0], SXH, kb, lane), hz);
        hn = wmma_bf16(a, ldsB(&Whh_s[2 * HP + j * 16][0], SXH, kb, lane), hn);
      }
      const int mb = mt * 16 + ((lane >> 4) * 8);
#pragma unroll
      for (int q = 0; q < 8; ++q) {
        const float r    = sigf(ir[q] + hr[q]);
        const float z    = sigf(iz[q] + hz[q]);
        const float n    = tanhf(in_[q] + r * hn[q]);
        const float hold = Hf[mb + q][nc];
        const float hnew = (1.0f - z) * n + z * hold;
        Hf[mb + q][nc] = hnew;
        if (nc < HID) {
          const size_t gi = ((rowBase + (mb + q)) * T + t) * HID + nc;
          out[gi] = hnew + resid[gi];
        }
      }
    }
    __syncthreads();
  }
}

// ---------------------------------------------------------------------------
// b3: hidden size 1, 3 fused layers; fully independent per (head, batch).
// ---------------------------------------------------------------------------
__global__ __launch_bounds__(128) void b3_block_kernel(
    const float* __restrict__ xin, float* __restrict__ out,
    const float* lb, const float* lw,
    const float* bhh0, const float* bih0, const float* whh0, const float* wih0,
    const float* bhh1, const float* bih1, const float* whh1, const float* wih1,
    const float* bhh2, const float* bih2, const float* whh2, const float* wih2) {
  const int h = blockIdx.x;
  const int b = threadIdx.x;
  float LW[8], WI0[3][8], WH0[3], BI0[3], BH0[3];
  float WI1[3], WH1[3], BI1[3], BH1[3];
  float WI2[3], WH2[3], BI2[3], BH2[3];
  const float LB = lb[h];
  for (int c = 0; c < 8; ++c) LW[c] = lw[h * 8 + c];
  for (int g = 0; g < 3; ++g) {
    for (int c = 0; c < 8; ++c) WI0[g][c] = wih0[(h * 3 + g) * 8 + c];
    WH0[g] = whh0[h * 3 + g]; BI0[g] = bih0[h * 3 + g]; BH0[g] = bhh0[h * 3 + g];
    WI1[g] = wih1[h * 3 + g]; WH1[g] = whh1[h * 3 + g];
    BI1[g] = bih1[h * 3 + g]; BH1[g] = bhh1[h * 3 + g];
    WI2[g] = wih2[h * 3 + g]; WH2[g] = whh2[h * 3 + g];
    BI2[g] = bih2[h * 3 + g]; BH2[g] = bhh2[h * 3 + g];
  }
  float h0 = 0.0f, h1 = 0.0f, h2 = 0.0f;
  const float* xr = xin + ((size_t)h * 128 + b) * 512 * 8;
  float* orow = out + ((size_t)h * 128 + b) * 512;
  for (int t = 0; t < 512; ++t) {
    const float4 xa = *(const float4*)(xr + t * 8);
    const float4 xb = *(const float4*)(xr + t * 8 + 4);
    const float x8[8] = {xa.x, xa.y, xa.z, xa.w, xb.x, xb.y, xb.z, xb.w};
    float res = LB;
    float g0 = BI0[0], g1 = BI0[1], g2 = BI0[2];
#pragma unroll
    for (int c = 0; c < 8; ++c) {
      const float xv = x8[c];
      res += xv * LW[c];
      g0 += xv * WI0[0][c]; g1 += xv * WI0[1][c]; g2 += xv * WI0[2][c];
    }
    {
      const float r = sigf(g0 + WH0[0] * h0 + BH0[0]);
      const float z = sigf(g1 + WH0[1] * h0 + BH0[1]);
      const float n = tanhf(g2 + r * (WH0[2] * h0 + BH0[2]));
      h0 = (1.0f - z) * n + z * h0;
    }
    const float x1 = h0 + res;
    {
      const float r = sigf(WI1[0] * x1 + BI1[0] + WH1[0] * h1 + BH1[0]);
      const float z = sigf(WI1[1] * x1 + BI1[1] + WH1[1] * h1 + BH1[1]);
      const float n = tanhf(WI1[2] * x1 + BI1[2] + r * (WH1[2] * h1 + BH1[2]));
      h1 = (1.0f - z) * n + z * h1;
    }
    const float x2 = h1 + x1;
    {
      const float r = sigf(WI2[0] * x2 + BI2[0] + WH2[0] * h2 + BH2[0]);
      const float z = sigf(WI2[1] * x2 + BI2[1] + WH2[1] * h2 + BH2[1]);
      const float n = tanhf(WI2[2] * x2 + BI2[2] + r * (WH2[2] * h2 + BH2[2]));
      h2 = (1.0f - z) * n + z * h2;
    }
    orow[t] = h2 + x2;
  }
}

// ---------------------------------------------------------------------------
// Helpers: weight repack (f32 -> zero-padded bf16), transpose, gather.
// ---------------------------------------------------------------------------
__global__ void repack_w_kernel(const float* __restrict__ src, __bf16* __restrict__ dst,
                                int heads, int HID, int HP, int Ks, int KP) {
  const long total = (long)heads * 3 * HP * KP;
  for (long i = (long)blockIdx.x * blockDim.x + threadIdx.x; i < total;
       i += (long)gridDim.x * blockDim.x) {
    const int k = (int)(i % KP);
    long r = i / KP;
    const int j = (int)(r % HP);
    r /= HP;
    const int g = (int)(r % 3);
    const int h = (int)(r / 3);
    float v = 0.0f;
    if (j < HID && k < Ks) v = src[(((long)h * 3 + g) * HID + j) * Ks + k];
    dst[i] = (__bf16)v;
  }
}

__global__ void repack_b_kernel(const float* __restrict__ src, float* __restrict__ dst,
                                int heads, int HID, int HP) {
  const int total = heads * 3 * HP;
  for (int i = blockIdx.x * blockDim.x + threadIdx.x; i < total;
       i += gridDim.x * blockDim.x) {
    const int j = i % HP;
    const int r = i / HP;
    const int g = r % 3;
    const int h = r / 3;
    dst[i] = (j < HID) ? src[(h * 3 + g) * HID + j] : 0.0f;
  }
}

// x[B=128][DM=8][S=64][L=512] -> y[DM][B][L][S], LDS-tiled 64x64 transpose.
__global__ __launch_bounds__(256) void transpose_x_kernel(
    const float* __restrict__ x, float* __restrict__ y) {
  __shared__ float tile[64][65];
  const int blk = blockIdx.x;  // ((b*8 + h) * 8 + tt)
  const int tt = blk & 7;
  const int bh = blk >> 3;
  const int h = bh & 7, b = bh >> 3;
  const int t0 = tt * 64;
  const float* src = x + (((size_t)b * 8 + h) * 64) * 512;  // [c][t]
  for (int i = threadIdx.x; i < 64 * 64; i += 256) {
    const int c = i >> 6, t = i & 63;  // consecutive tid -> consecutive t
    tile[t][c] = src[(size_t)c * 512 + t0 + t];
  }
  __syncthreads();
  float* dst = y + (((size_t)h * 128 + b) * 512 + t0) * 64;  // [t][c]
  for (int i = threadIdx.x; i < 64 * 64; i += 256) {
    const int t = i >> 6, c = i & 63;  // consecutive tid -> consecutive c
    dst[(size_t)t * 64 + c] = tile[t][c];
  }
}

// g4[b][t][h] = o3[h][b][t]
__global__ void gather_heads_kernel(const float* __restrict__ o3, float* __restrict__ g,
                                    int total) {
  for (int i = blockIdx.x * blockDim.x + threadIdx.x; i < total;
       i += gridDim.x * blockDim.x) {
    const int h = i & 7;
    const int t = (i >> 3) & 511;
    const int b = i >> 12;
    g[i] = o3[(((long)h * 128 + b) * 512) + t];
  }
}

// ---------------------------------------------------------------------------
// Host orchestration.
// ---------------------------------------------------------------------------
extern "C" void kernel_launch(void* const* d_in, const int* in_sizes, int n_in,
                              void* d_out, int out_size, void* d_ws, size_t ws_size,
                              hipStream_t stream) {
  (void)n_in; (void)out_size; (void)ws_size;
  auto F = [&](int i) { return (const float*)d_in[i]; };

  struct BP { const float *lw, *lb, *wih[3], *whh[3], *bih[3], *bhh[3]; } bp[5];
  const float *fcw[4], *fcb[4];
  const float* X;

  const bool insertion = (in_sizes[0] > 1000000);  // x first => insertion order
  if (insertion) {
    X = F(0);
    int idx = 1;
    for (int blk = 0; blk < 5; ++blk) {
      bp[blk].lw = F(idx++); bp[blk].lb = F(idx++);
      for (int l = 0; l < 3; ++l) {
        bp[blk].wih[l] = F(idx++); bp[blk].whh[l] = F(idx++);
        bp[blk].bih[l] = F(idx++); bp[blk].bhh[l] = F(idx++);
      }
    }
    for (int k = 0; k < 4; ++k) { fcw[k] = F(idx++); fcb[k] = F(idx++); }
  } else {  // JAX sorted-dict flattening: params (sorted keys), then x
    for (int blk = 0; blk < 5; ++blk) {
      const int base = blk * 14;
      for (int l = 0; l < 3; ++l) {
        bp[blk].bhh[l] = F(base + 4 * l + 0);
        bp[blk].bih[l] = F(base + 4 * l + 1);
        bp[blk].whh[l] = F(base + 4 * l + 2);
        bp[blk].wih[l] = F(base + 4 * l + 3);
      }
      bp[blk].lb = F(base + 12);
      bp[blk].lw = F(base + 13);
    }
    for (int k = 0; k < 4; ++k) { fcb[k] = F(70 + 2 * k); fcw[k] = F(70 + 2 * k + 1); }
    X = F(78);
  }

  // Workspace layout: [X0 : 128MB][arena : 128MB][packed weights : ~2MB]
  const size_t BIG = 33554432;  // floats = 8*128*512*64
  float* X0 = (float*)d_ws;
  float* AR = X0 + BIG;
  float* R0 = AR;
  float* r2 = AR;
  float* o2a = AR + (size_t)8 * 128 * 512 * 8;  // +4,194,304
  float* o3  = o2a + 4194304;
  float* g4  = o3 + 524288;
  float* r4  = g4 + 524288;
  float* o4  = r4 + 1048576;
  float* r5  = o4 + 1048576;
  float* o5  = r5 + 524288;
  float* f1  = o5 + 524288;
  float* f2  = f1 + 128 * 256;
  float* f3  = f2 + 128 * 128;

  char* pk = (char*)(X0 + 2 * BIG);
  auto palloc = [&](size_t bytes) {
    void* r = (void*)pk;
    pk += (bytes + 255) & ~(size_t)255;
    return r;
  };

  struct PK { __bf16* wih; __bf16* whh; float* bih; float* bhh; };
  auto pack = [&](const float* wih, const float* whh, const float* bih,
                  const float* bhh, int heads, int HID, int HP, int IN, int KI,
                  int KH) {
    PK p;
    p.wih = (__bf16*)palloc((size_t)heads * 3 * HP * KI * 2);
    p.whh = (__bf16*)palloc((size_t)heads * 3 * HP * KH * 2);
    p.bih = (float*)palloc((size_t)heads * 3 * HP * 4);
    p.bhh = (float*)palloc((size_t)heads * 3 * HP * 4);
    const long tw = (long)heads * 3 * HP * KI;
    repack_w_kernel<<<(int)((tw + 255) / 256), 256, 0, stream>>>(wih, p.wih, heads,
                                                                 HID, HP, IN, KI);
    const long th = (long)heads * 3 * HP * KH;
    repack_w_kernel<<<(int)((th + 255) / 256), 256, 0, stream>>>(whh, p.whh, heads,
                                                                 HID, HP, HID, KH);
    const int tb = heads * 3 * HP;
    repack_b_kernel<<<(tb + 255) / 256, 256, 0, stream>>>(bih, p.bih, heads, HID, HP);
    repack_b_kernel<<<(tb + 255) / 256, 256, 0, stream>>>(bhh, p.bhh, heads, HID, HP);
    return p;
  };

  auto gemm = [&](const float* A, const float* W, const float* bias, float* C, int G,
                  int M, int N, int K, size_t sA, size_t sW, size_t sB, size_t sC,
                  int relu) {
    const int MT = (M + 15) / 16, NT = (N + 15) / 16;
    const long tiles = (long)G * MT * NT;
    const int blocks = (int)((tiles + 7) / 8);
    const bool ka = (K % 32) == 0;
    if (ka && relu)
      wmma_gemm_kernel<true, true><<<blocks, 256, 0, stream>>>(
          A, W, bias, C, M, N, K, MT, NT, sA, sW, sB, sC, tiles);
    else if (ka)
      wmma_gemm_kernel<true, false><<<blocks, 256, 0, stream>>>(
          A, W, bias, C, M, N, K, MT, NT, sA, sW, sB, sC, tiles);
    else if (relu)
      wmma_gemm_kernel<false, true><<<blocks, 256, 0, stream>>>(
          A, W, bias, C, M, N, K, MT, NT, sA, sW, sB, sC, tiles);
    else
      wmma_gemm_kernel<false, false><<<blocks, 256, 0, stream>>>(
          A, W, bias, C, M, N, K, MT, NT, sA, sW, sB, sC, tiles);
  };

  // --- weight repacking (bf16, padded) ---
  PK p1[3], p2[3], p4[3], p5[3];
  for (int l = 0; l < 3; ++l)
    p1[l] = pack(bp[0].wih[l], bp[0].whh[l], bp[0].bih[l], bp[0].bhh[l], 8, 64, 64,
                 64, 64, 64);
  p2[0] = pack(bp[1].wih[0], bp[1].whh[0], bp[1].bih[0], bp[1].bhh[0], 8, 8, 16, 64,
               64, 32);
  for (int l = 1; l < 3; ++l)
    p2[l] = pack(bp[1].wih[l], bp[1].whh[l], bp[1].bih[l], bp[1].bhh[l], 8, 8, 16, 8,
                 32, 32);
  p4[0] = pack(bp[3].wih[0], bp[3].whh[0], bp[3].bih[0], bp[3].bhh[0], 1, 16, 16, 8,
               32, 32);
  for (int l = 1; l < 3; ++l)
    p4[l] = pack(bp[3].wih[l], bp[3].whh[l], bp[3].bih[l], bp[3].bhh[l], 1, 16, 16,
                 16, 32, 32);
  p5[0] = pack(bp[4].wih[0], bp[4].whh[0], bp[4].bih[0], bp[4].bhh[0], 1, 8, 16, 16,
               32, 32);
  for (int l = 1; l < 3; ++l)
    p5[l] = pack(bp[4].wih[l], bp[4].whh[l], bp[4].bih[l], bp[4].bhh[l], 1, 8, 16, 8,
                 32, 32);

  // --- stage input: [B,DM,S,L] -> [DM][B][L][S] ---
  transpose_x_kernel<<<8192, 256, 0, stream>>>(X, X0);

  // --- b1: heads=8, in=64, hid=64 ---
  gemm(X0, bp[0].lw, bp[0].lb, R0, 8, 65536, 64, 64, 4194304, 4096, 64, 4194304, 0);
  gru_scan_kernel<64, 64><<<8, 256, 0, stream>>>(X0, R0, X0, p1[0].wih, p1[0].whh,
                                                 p1[0].bih, p1[0].bhh);
  gru_scan_kernel<64, 64><<<8, 256, 0, stream>>>(X0, X0, R0, p1[1].wih, p1[1].whh,
                                                 p1[1].bih, p1[1].bhh);
  gru_scan_kernel<64, 64><<<8, 256, 0, stream>>>(R0, R0, X0, p1[2].wih, p1[2].whh,
                                                 p1[2].bih, p1[2].bhh);

  // --- b2: heads=8, in=64, hid=8 ---
  gemm(X0, bp[1].lw, bp[1].lb, r2, 8, 65536, 8, 64, 4194304, 512, 8, 524288, 0);
  gru_scan_kernel<64, 8><<<8, 256, 0, stream>>>(X0, r2, o2a, p2[0].wih, p2[0].whh,
                                                p2[0].bih, p2[0].bhh);
  gru_scan_kernel<8, 8><<<8, 256, 0, stream>>>(o2a, o2a, r2, p2[1].wih, p2[1].whh,
                                               p2[1].bih, p2[1].bhh);
  gru_scan_kernel<8, 8><<<8, 256, 0, stream>>>(r2, r2, o2a, p2[2].wih, p2[2].whh,
                                               p2[2].bih, p2[2].bhh);

  // --- b3: heads=8, in=8, hid=1 (scalar recurrence, 3 fused layers) ---
  b3_block_kernel<<<8, 128, 0, stream>>>(
      o2a, o3, bp[2].lb, bp[2].lw, bp[2].bhh[0], bp[2].bih[0], bp[2].whh[0],
      bp[2].wih[0], bp[2].bhh[1], bp[2].bih[1], bp[2].whh[1], bp[2].wih[1],
      bp[2].bhh[2], bp[2].bih[2], bp[2].whh[2], bp[2].wih[2]);

  // --- concat heads: [B][L][DM=8] ---
  gather_heads_kernel<<<2048, 256, 0, stream>>>(o3, g4, 524288);

  // --- b4: in=8, hid=16 ---
  gemm(g4, bp[3].lw, bp[3].lb, r4, 1, 65536, 16, 8, 0, 0, 0, 0, 0);
  gru_scan_kernel<8, 16><<<1, 256, 0, stream>>>(g4, r4, o4, p4[0].wih, p4[0].whh,
                                                p4[0].bih, p4[0].bhh);
  gru_scan_kernel<16, 16><<<1, 256, 0, stream>>>(o4, o4, r4, p4[1].wih, p4[1].whh,
                                                 p4[1].bih, p4[1].bhh);
  gru_scan_kernel<16, 16><<<1, 256, 0, stream>>>(r4, r4, o4, p4[2].wih, p4[2].whh,
                                                 p4[2].bih, p4[2].bhh);

  // --- b5: in=16, hid=8 ---
  gemm(o4, bp[4].lw, bp[4].lb, r5, 1, 65536, 8, 16, 0, 0, 0, 0, 0);
  gru_scan_kernel<16, 8><<<1, 256, 0, stream>>>(o4, r5, o5, p5[0].wih, p5[0].whh,
                                                p5[0].bih, p5[0].bhh);
  gru_scan_kernel<8, 8><<<1, 256, 0, stream>>>(o5, o5, r5, p5[1].wih, p5[1].whh,
                                               p5[1].bih, p5[1].bhh);
  gru_scan_kernel<8, 8><<<1, 256, 0, stream>>>(r5, r5, o5, p5[2].wih, p5[2].whh,
                                               p5[2].bih, p5[2].bhh);

  // --- FC head: [128,4096] -> 256 -> 128 -> 64 -> 20, relu each ---
  gemm(o5, fcw[0], fcb[0], f1, 1, 128, 256, 4096, 0, 0, 0, 0, 1);
  gemm(f1, fcw[1], fcb[1], f2, 1, 128, 128, 256, 0, 0, 0, 0, 1);
  gemm(f2, fcw[2], fcb[2], f3, 1, 128, 64, 128, 0, 0, 0, 0, 1);
  gemm(f3, fcw[3], fcb[3], (float*)d_out, 1, 128, 20, 64, 0, 0, 0, 0, 1);
}
